// AdditiveAttention_21921513079340
// MI455X (gfx1250) — matrix operand order repaired and checked by reading, not compiled, not run on hardware
//
#include <hip/hip_runtime.h>
#include <math.h>

// ---------------------------------------------------------------------------
// Additive (Bahdanau) attention for MI455X (gfx1250, wave32, WMMA).
//   B=32, S=2048, D=1024
// k = keys @ Wk^T (137 GFLOP) -> bf16 WMMA with fused tanh/v-dot epilogue
// (k never hits memory). 4x4 register tiling: 16 WMMAs per K-step, 0.5 KB
// fragment traffic per WMMA. Context pass uses GLOBAL_LOAD_ASYNC_TO_LDS
// double buffering (ASYNCcnt).
// ---------------------------------------------------------------------------

typedef __attribute__((ext_vector_type(16))) __bf16 bf16x16;
typedef __attribute__((ext_vector_type(8)))  float  f32x8;
typedef __attribute__((ext_vector_type(4)))  int    v4i;

struct U32x8 { uint4 lo, hi; };   // 32 bytes == one bf16x16 fragment

#define D_MODEL 1024
#define SEQ     2048
#define BATCH   32
#define ROWS_PER_WG 64          // keys tile rows per workgroup
#define LDS_PITCH   1032        // 1024 + 8 ushorts: de-conflict 2KB-strided rows

#if defined(__has_builtin)
#if __has_builtin(__builtin_amdgcn_global_load_async_to_lds_b128)
#define HAVE_ASYNC_LDS 1
#endif
#endif
#ifndef HAVE_ASYNC_LDS
#define HAVE_ASYNC_LDS 0
#endif

#if defined(__has_builtin)
#if __has_builtin(__builtin_amdgcn_s_wait_asynccnt)
#define WAIT_ASYNC(n) __builtin_amdgcn_s_wait_asynccnt(n)
#endif
#endif
#ifndef WAIT_ASYNC
#define WAIT_ASYNC(n) asm volatile("s_wait_asynccnt %0" ::"i"(n) : "memory")
#endif

__device__ __forceinline__ unsigned short f32_to_bf16(float f) {
  unsigned u = __float_as_uint(f);
  unsigned r = u + 0x7fffu + ((u >> 16) & 1u);   // round-to-nearest-even
  return (unsigned short)(r >> 16);
}

__device__ __forceinline__ float fast_tanh(float x) {
#if defined(__has_builtin)
#if __has_builtin(__builtin_amdgcn_tanhf)
  return __builtin_amdgcn_tanhf(x);     // V_TANH_F32 (CDNA5 trans op)
#else
  return tanhf(x);
#endif
#else
  return tanhf(x);
#endif
}

// ---------------------------------------------------------------------------
// Kernel 1: q = query @ Wq^T + Wq_b   [32,1024] (tiny, fp32 VALU)
// ---------------------------------------------------------------------------
__global__ __launch_bounds__(256)
void qproj_kernel(const float* __restrict__ query,
                  const float* __restrict__ Wq_w,
                  const float* __restrict__ Wq_b,
                  float* __restrict__ qbuf) {
  int gid = blockIdx.x * 256 + threadIdx.x;     // 32768 outputs
  int b = gid >> 10, n = gid & 1023;
  const float4* qr = (const float4*)(query + (size_t)b * D_MODEL);
  const float4* wr = (const float4*)(Wq_w + (size_t)n * D_MODEL);
  float acc = 0.f;
  for (int i = 0; i < D_MODEL / 4; ++i) {
    float4 a = qr[i], w = wr[i];
    acc += a.x * w.x + a.y * w.y + a.z * w.z + a.w * w.w;
  }
  qbuf[gid] = acc + Wq_b[n];
}

// ---------------------------------------------------------------------------
// Kernel 2: convert Wk (row-major [n][d]) fp32 -> bf16 (2 MB, L2-resident)
// ---------------------------------------------------------------------------
__global__ __launch_bounds__(256)
void wkconv_kernel(const float* __restrict__ Wk_w,
                   unsigned short* __restrict__ WkB) {
  int gid4 = blockIdx.x * 256 + threadIdx.x;    // 262144 float4
  float4 v = ((const float4*)Wk_w)[gid4];
  uint2 p;
  p.x = (unsigned)f32_to_bf16(v.x) | ((unsigned)f32_to_bf16(v.y) << 16);
  p.y = (unsigned)f32_to_bf16(v.z) | ((unsigned)f32_to_bf16(v.w) << 16);
  ((uint2*)WkB)[gid4] = p;
}

// ---------------------------------------------------------------------------
// Kernel 3: scores[r] = v_b + sum_n v_w[n] * tanh(q[b,n] + Wk_b[n] + k[r,n])
// 128 threads / 4 waves per WG; each wave: all 64 rows x its 256-col quarter,
// 4x4 register tile -> 16 WMMAs per 32-wide K step.
// ---------------------------------------------------------------------------
__global__ __launch_bounds__(128)
void scores_kernel(const float* __restrict__ keys,
                   const unsigned short* __restrict__ WkB,
                   const float* __restrict__ qbuf,
                   const float* __restrict__ Wk_b,
                   const float* __restrict__ v_w,
                   const float* __restrict__ v_b,
                   float* __restrict__ scores) {
  __shared__ unsigned short sKeys[ROWS_PER_WG * LDS_PITCH];  // ~129 KB bf16
  __shared__ float sQ[D_MODEL];                              // 4 KB
  __shared__ float sP[4][ROWS_PER_WG];                       // 1 KB

  const int tid   = threadIdx.x;
  const int wave  = tid >> 5;          // 0..3 -> N quarter
  const int lane  = tid & 31;
  const int l16   = lane & 15;
  const int khalf = lane >> 4;

  const int row_base = blockIdx.x * ROWS_PER_WG;   // aligned within a batch
  const int b = row_base / SEQ;

  // -- stage keys tile [64 x 1024] fp32 -> bf16 LDS (padded pitch) --
  const float4* kt = (const float4*)(keys + (size_t)row_base * D_MODEL);
  for (int i = 0; i < (ROWS_PER_WG * D_MODEL) / (4 * 128); ++i) {  // 128 iters
    int e4 = i * 128 + tid;
    int r  = e4 >> 8;                // 256 float4 per row
    int c4 = e4 & 255;
    float4 v = kt[r * 256 + c4];
    uint2 p;
    p.x = (unsigned)f32_to_bf16(v.x) | ((unsigned)f32_to_bf16(v.y) << 16);
    p.y = (unsigned)f32_to_bf16(v.z) | ((unsigned)f32_to_bf16(v.w) << 16);
    *(uint2*)&sKeys[r * LDS_PITCH + c4 * 4] = p;
  }
  for (int i = tid; i < D_MODEL; i += 128) sQ[i] = qbuf[b * D_MODEL + i];
  __syncthreads();

  float psum[4][8];
#pragma unroll
  for (int mi = 0; mi < 4; ++mi)
#pragma unroll
    for (int i = 0; i < 8; ++i) psum[mi][i] = 0.f;
  const float vb0 = v_b[0];
  const int nwave = wave * 256;

#pragma unroll 1
  for (int c = 0; c < 4; ++c) {                  // 4 chunks of 64 cols
    f32x8 acc[4][4];
#pragma unroll
    for (int mi = 0; mi < 4; ++mi)
#pragma unroll
      for (int ni = 0; ni < 4; ++ni) acc[mi][ni] = f32x8{};

#pragma unroll 1
    for (int k0 = 0; k0 < D_MODEL; k0 += 32) {
      // 4 B fragments (32x16 each): lane col n = l16, K span by lane half.
      bf16x16 bfrag[4];
#pragma unroll
      for (int ni = 0; ni < 4; ++ni) {
        int n = nwave + c * 64 + ni * 16 + l16;
        const uint4* bp =
            (const uint4*)(WkB + (size_t)n * D_MODEL + k0 + khalf * 16);
        U32x8 braw;
        braw.lo = bp[0];
        braw.hi = bp[1];
        bfrag[ni] = __builtin_bit_cast(bf16x16, braw);
      }
      // 4 A fragments (16x32): row M = lane%16, K-half = lane/16.
#pragma unroll
      for (int mi = 0; mi < 4; ++mi) {
        const unsigned short* aRow = &sKeys[(mi * 16 + l16) * LDS_PITCH];
        U32x8 araw;
        araw.lo = *(const uint4*)&aRow[k0 + khalf * 16];
        araw.hi = *(const uint4*)&aRow[k0 + khalf * 16 + 8];
        bf16x16 afrag = __builtin_bit_cast(bf16x16, araw);
#pragma unroll
        for (int ni = 0; ni < 4; ++ni)
          acc[mi][ni] = __builtin_amdgcn_wmma_f32_16x16x32_bf16(
              false, afrag, false, bfrag[ni], (short)0, acc[mi][ni], false,
              false);
      }
    }

    // fused epilogue: psum[row] += v_w[n] * tanh(q[n] + Wk_b[n] + k)
    // C layout: vgpr i, lanes 0-15 -> M=i, lanes 16-31 -> M=8+i; N = lane%16.
#pragma unroll
    for (int ni = 0; ni < 4; ++ni) {
      int n = nwave + c * 64 + ni * 16 + l16;
      float qn = sQ[n];
      float vw = v_w[n];
      float kb = Wk_b[n];
#pragma unroll
      for (int mi = 0; mi < 4; ++mi)
#pragma unroll
        for (int i = 0; i < 8; ++i)
          psum[mi][i] += vw * fast_tanh(qn + kb + acc[mi][ni][i]);
    }
  }

  // reduce across the 16 lanes of each half (cols of the C tiles)
#pragma unroll
  for (int m = 1; m <= 8; m <<= 1)
#pragma unroll
    for (int mi = 0; mi < 4; ++mi)
#pragma unroll
      for (int i = 0; i < 8; ++i)
        psum[mi][i] += __shfl_xor(psum[mi][i], m, 32);

  // cross-wave (cross-N-quarter) reduction via LDS
  if (l16 == 0) {
#pragma unroll
    for (int mi = 0; mi < 4; ++mi)
#pragma unroll
      for (int i = 0; i < 8; ++i)
        sP[wave][mi * 16 + khalf * 8 + i] = psum[mi][i];
  }
  __syncthreads();
  if (tid < ROWS_PER_WG) {
    float s = sP[0][tid] + sP[1][tid] + sP[2][tid] + sP[3][tid];
    scores[row_base + tid] = s + vb0;
  }
}

// ---------------------------------------------------------------------------
// Kernel 4: softmax over S per batch row -> attn (written straight to d_out)
// ---------------------------------------------------------------------------
__global__ __launch_bounds__(256)
void softmax_kernel(const float* __restrict__ scores,
                    float* __restrict__ attn) {
  __shared__ float red[256];
  int b = blockIdx.x, tid = threadIdx.x;
  float v[8];
#pragma unroll
  for (int i = 0; i < 8; ++i) v[i] = scores[b * SEQ + i * 256 + tid];
  float mx = v[0];
#pragma unroll
  for (int i = 1; i < 8; ++i) mx = fmaxf(mx, v[i]);
  red[tid] = mx;
  __syncthreads();
  for (int s = 128; s > 0; s >>= 1) {
    if (tid < s) red[tid] = fmaxf(red[tid], red[tid + s]);
    __syncthreads();
  }
  mx = red[0];
  __syncthreads();
  float sum = 0.f;
#pragma unroll
  for (int i = 0; i < 8; ++i) {
    v[i] = __expf(v[i] - mx);
    sum += v[i];
  }
  red[tid] = sum;
  __syncthreads();
  for (int s = 128; s > 0; s >>= 1) {
    if (tid < s) red[tid] += red[tid + s];
    __syncthreads();
  }
  float inv = 1.f / red[0];
#pragma unroll
  for (int i = 0; i < 8; ++i) attn[b * SEQ + i * 256 + tid] = v[i] * inv;
}

// ---------------------------------------------------------------------------
// Kernel 5: context partials = attn @ keys over 256-row s-chunks.
// Async path: 8-row (32 KB) groups copied memory->LDS with
// GLOBAL_LOAD_ASYNC_TO_LDS_B128, double-buffered on ASYNCcnt.
// ---------------------------------------------------------------------------
#if HAVE_ASYNC_LDS
typedef __attribute__((address_space(1))) v4i gv4i;   // global int4
typedef __attribute__((address_space(3))) v4i lv4i;   // LDS int4

__global__ __launch_bounds__(256)
void ctx_partial_kernel(const float* __restrict__ keys,
                        const float* __restrict__ attn,
                        float* __restrict__ partials) {
  __shared__ float buf[2][8 * D_MODEL];          // 2 x 32 KB
  int b = blockIdx.y, sc = blockIdx.x, tid = threadIdx.x;
  const float* gbase = keys + ((size_t)b * SEQ + sc * 256) * D_MODEL;
  const float* ap = attn + b * SEQ + sc * 256;

  // copy one 8-row group (2048 float4) into LDS slot: 8 async b128 per thread
#define ISSUE_GROUP(g, slot)                                                  \
  {                                                                           \
    _Pragma("unroll") for (int t = 0; t < 8; ++t) {                           \
      int idx4 = t * 256 + tid;                                               \
      const float* src = gbase + (size_t)(g)*8192 + (size_t)idx4 * 4;         \
      float* dst = &buf[slot][idx4 * 4];                                      \
      __builtin_amdgcn_global_load_async_to_lds_b128(                         \
          (gv4i*)(uintptr_t)src, (lv4i*)(unsigned)(uintptr_t)dst, 0, 0);      \
    }                                                                         \
  }

  ISSUE_GROUP(0, 0);
  float4 acc = {0.f, 0.f, 0.f, 0.f};
#pragma unroll 1
  for (int g = 0; g < 32; ++g) {
    if (g + 1 < 32) {
      ISSUE_GROUP(g + 1, (g + 1) & 1);
      WAIT_ASYNC(8);          // group g complete (async loads finish in order)
    } else {
      WAIT_ASYNC(0);
    }
    __syncthreads();
    const float4* kb = (const float4*)&buf[g & 1][0] + tid;
#pragma unroll
    for (int r = 0; r < 8; ++r) {
      float a = ap[g * 8 + r];
      float4 k = kb[r * 256];
      acc.x += a * k.x; acc.y += a * k.y; acc.z += a * k.z; acc.w += a * k.w;
    }
    __syncthreads();          // group g buffer free for reuse
  }
  ((float4*)partials)[(b * 8 + sc) * 256 + tid] = acc;
#undef ISSUE_GROUP
}
#else
__global__ __launch_bounds__(256)
void ctx_partial_kernel(const float* __restrict__ keys,
                        const float* __restrict__ attn,
                        float* __restrict__ partials) {
  int b = blockIdx.y, sc = blockIdx.x, tid = threadIdx.x;
  const float4* kp =
      (const float4*)(keys + ((size_t)b * SEQ + sc * 256) * D_MODEL) + tid;
  const float* ap = attn + b * SEQ + sc * 256;
  float4 acc = {0.f, 0.f, 0.f, 0.f};
  for (int s = 0; s < 256; ++s) {
    float a = ap[s];
    float4 k = kp[(size_t)s * 256];
    acc.x += a * k.x; acc.y += a * k.y; acc.z += a * k.z; acc.w += a * k.w;
  }
  ((float4*)partials)[(b * 8 + sc) * 256 + tid] = acc;
}
#endif

__global__ __launch_bounds__(256)
void ctx_reduce_kernel(const float* __restrict__ partials,
                       float* __restrict__ ctx) {
  int e = blockIdx.x * 256 + threadIdx.x;       // 8192 float4
  int b = e >> 8, d4 = e & 255;
  float4 s = {0.f, 0.f, 0.f, 0.f};
#pragma unroll
  for (int i = 0; i < 8; ++i) {
    float4 p = ((const float4*)partials)[(b * 8 + i) * 256 + d4];
    s.x += p.x; s.y += p.y; s.z += p.z; s.w += p.w;
  }
  ((float4*)ctx)[e] = s;
}

// ---------------------------------------------------------------------------
extern "C" void kernel_launch(void* const* d_in, const int* in_sizes, int n_in,
                              void* d_out, int out_size, void* d_ws,
                              size_t ws_size, hipStream_t stream) {
  (void)in_sizes; (void)n_in; (void)out_size; (void)ws_size;
  const float* query = (const float*)d_in[0];
  const float* keys  = (const float*)d_in[1];
  const float* Wq_w  = (const float*)d_in[2];
  const float* Wq_b  = (const float*)d_in[3];
  const float* Wk_w  = (const float*)d_in[4];
  const float* Wk_b  = (const float*)d_in[5];
  const float* v_w   = (const float*)d_in[6];
  const float* v_b   = (const float*)d_in[7];

  float* out      = (float*)d_out;
  float* ctx_out  = out;                       // [32,1024]
  float* attn_out = out + BATCH * D_MODEL;     // [32,2048]

  // workspace layout (all 256B aligned): q | WkB(bf16) | scores | partials
  char* ws = (char*)d_ws;
  float*          qbuf     = (float*)ws;                              // 128 KB
  unsigned short* WkB      = (unsigned short*)(ws + 131072);          // 2 MB
  float*          scores   = (float*)(ws + 131072 + 2097152);         // 256 KB
  float*          partials = (float*)(ws + 131072 + 2097152 + 262144);// 1 MB

  qproj_kernel<<<(BATCH * D_MODEL) / 256, 256, 0, stream>>>(query, Wq_w, Wq_b,
                                                            qbuf);
  wkconv_kernel<<<(D_MODEL * D_MODEL) / (4 * 256), 256, 0, stream>>>(Wk_w, WkB);
  scores_kernel<<<(BATCH * SEQ) / ROWS_PER_WG, 128, 0, stream>>>(
      keys, WkB, qbuf, Wk_b, v_w, v_b, scores);
  softmax_kernel<<<BATCH, 256, 0, stream>>>(scores, attn_out);
  ctx_partial_kernel<<<dim3(8, BATCH), 256, 0, stream>>>(keys, attn_out,
                                                         partials);
  ctx_reduce_kernel<<<(BATCH * D_MODEL) / (4 * 256), 256, 0, stream>>>(partials,
                                                                       ctx_out);
}
